// DynamicConv1dTBC_60138132078899
// MI455X (gfx1250) — compile-verified, matrix operand order checked by reading, not run on hardware
//
#include <hip/hip_runtime.h>
#include <math.h>

typedef __attribute__((ext_vector_type(16))) _Float16 v16h;
typedef __attribute__((ext_vector_type(8)))  float    v8f;

#define T_DIM   2048
#define B_DIM   8
#define C_DIM   1024
#define H_DIM   16
#define K_CONV  7
#define M_DIM   (T_DIM * B_DIM)   /* 16384 rows */
#define NCAT    144               /* 2H + H*K = 32 + 112 */

#if __has_builtin(__builtin_amdgcn_global_load_async_to_lds_b128)
#define HAVE_ASYNC_LDS 1
typedef int v4i_vec __attribute__((vector_size(16)));
typedef __attribute__((address_space(1))) v4i_vec* gptr4_t;
typedef __attribute__((address_space(3))) v4i_vec* lptr4_t;
#endif

__device__ __forceinline__ void wait_async0() {
#if __has_builtin(__builtin_amdgcn_s_wait_asynccnt)
  __builtin_amdgcn_s_wait_asynccnt(0);
#else
  asm volatile("s_wait_asynccnt 0x0" ::: "memory");
#endif
}

// ---------------------------------------------------------------------------
// fp32 -> fp16 conversion, 4 elements per thread
// ---------------------------------------------------------------------------
__global__ void cvt_f32_to_f16(const float* __restrict__ src,
                               _Float16* __restrict__ dst, int n4) {
  int i = blockIdx.x * blockDim.x + threadIdx.x;
  if (i >= n4) return;
  float4 v = ((const float4*)src)[i];
  union { _Float16 h[4]; uint2 u; } o;
  o.h[0] = (_Float16)v.x; o.h[1] = (_Float16)v.y;
  o.h[2] = (_Float16)v.z; o.h[3] = (_Float16)v.w;
  *(uint2*)(dst + (size_t)i * 4) = o.u;
}

// ---------------------------------------------------------------------------
// Build zero-padded 256x1024 f16 weight block: rows 0..31 = glu_w,
// rows 32..143 = w_weight, rows 144..255 = 0
// ---------------------------------------------------------------------------
__global__ void prep_wcat(const float* __restrict__ glu_w,
                          const float* __restrict__ w_weight,
                          _Float16* __restrict__ wcat) {
  int i = blockIdx.x * blockDim.x + threadIdx.x;   // over 256*1024/4 chunks
  int base = i * 4;
  int row = base >> 10;
  int col = base & 1023;
  float4 v = make_float4(0.f, 0.f, 0.f, 0.f);
  if (row < 32)        v = *(const float4*)&glu_w[(size_t)row * C_DIM + col];
  else if (row < NCAT) v = *(const float4*)&w_weight[(size_t)(row - 32) * C_DIM + col];
  union { _Float16 h[4]; uint2 u; } o;
  o.h[0] = (_Float16)v.x; o.h[1] = (_Float16)v.y;
  o.h[2] = (_Float16)v.z; o.h[3] = (_Float16)v.w;
  *(uint2*)(wcat + (size_t)base) = o.u;
}

// ---------------------------------------------------------------------------
// Tiled WMMA GEMM: C[M x N] = A[M x K] * B[N x K]^T (+ bias)
//   Block = 256 threads = 8 waves; block tile 128(M) x 128(N), K-step 32.
//   Waves arranged 4(M) x 2(N); each wave computes 32 x 64 via 2x4 WMMA
//   accumulators. LDS double-buffered, one barrier per K-step.
//   Next tile is moved global->LDS with GLOBAL_LOAD_ASYNC_TO_LDS_B128
//   (ASYNCcnt-tracked, no VGPR round trip); s_wait_asynccnt 0 sits after
//   the tile's WMMAs so HBM latency hides behind compute.
// ---------------------------------------------------------------------------
__global__ void __launch_bounds__(256)
gemm_f16(const _Float16* __restrict__ A,
         const _Float16* __restrict__ Bw,
         float* __restrict__ Cout,
         const float* __restrict__ bias,
         int N, int Kd, int ldc) {
  // padded row stride 40 halves (80B) keeps 16B ds loads conflict-light
  __shared__ __align__(16) _Float16 As[2][128 * 40];
  __shared__ __align__(16) _Float16 Bs[2][128 * 40];

  const int tid   = threadIdx.x;
  const int lane  = tid & 31;
  const int wave  = tid >> 5;
  const int wm    = wave & 3;        // M quarter (32 rows each)
  const int wn    = wave >> 2;       // N half (64 cols each)
  const int laneM = lane & 15;
  const int kh    = lane >> 4;       // K-half selector per ISA 16-bit layout

  const int m0 = blockIdx.y * 128;
  const int n0 = blockIdx.x * 128;

  v8f acc[2][4];
  const v8f vzero = {0.f, 0.f, 0.f, 0.f, 0.f, 0.f, 0.f, 0.f};
#pragma unroll
  for (int mt = 0; mt < 2; ++mt)
#pragma unroll
    for (int nt = 0; nt < 4; ++nt) acc[mt][nt] = vzero;

  // global->LDS staging: 128 rows x 32 halves per matrix; 256 threads,
  // each moves 2x16B of A and 2x16B of B
  const int srow = tid >> 1;
  const int skc  = (tid & 1) * 16;   // halves
  const int sidx = srow * 40 + skc;  // LDS destination (halves)

  const int nk = Kd >> 5;            // K-steps of 32

#ifdef HAVE_ASYNC_LDS
  // -------- async global->LDS double-buffer pipeline ----------------------
  auto stage = [&](int buf, int k0) {
    const _Float16* as = &A[(size_t)(m0 + srow) * Kd + k0 + skc];
    const _Float16* bs = &Bw[(size_t)(n0 + srow) * Kd + k0 + skc];
    // imm offset applies to BOTH global and LDS addresses (ISA 15.18)
    __builtin_amdgcn_global_load_async_to_lds_b128(
        (gptr4_t)as, (lptr4_t)&As[buf][sidx], 0, 0);
    __builtin_amdgcn_global_load_async_to_lds_b128(
        (gptr4_t)as, (lptr4_t)&As[buf][sidx], 16, 0);
    __builtin_amdgcn_global_load_async_to_lds_b128(
        (gptr4_t)bs, (lptr4_t)&Bs[buf][sidx], 0, 0);
    __builtin_amdgcn_global_load_async_to_lds_b128(
        (gptr4_t)bs, (lptr4_t)&Bs[buf][sidx], 16, 0);
  };
  stage(0, 0);
  wait_async0();
  __syncthreads();
#else
  // -------- register-staged double-buffer pipeline ------------------------
  uint4 ra0, ra1, rb0, rb1;
  auto fetch = [&](int k0) {
    const _Float16* as = &A[(size_t)(m0 + srow) * Kd + k0 + skc];
    const _Float16* bs = &Bw[(size_t)(n0 + srow) * Kd + k0 + skc];
    ra0 = *(const uint4*)as;  ra1 = *(const uint4*)(as + 8);
    rb0 = *(const uint4*)bs;  rb1 = *(const uint4*)(bs + 8);
  };
  auto store = [&](int buf) {
    *(uint4*)&As[buf][sidx]     = ra0;
    *(uint4*)&As[buf][sidx + 8] = ra1;
    *(uint4*)&Bs[buf][sidx]     = rb0;
    *(uint4*)&Bs[buf][sidx + 8] = rb1;
  };
  fetch(0);
  store(0);
  __syncthreads();
#endif

  for (int i = 0; i < nk; ++i) {
    const int cur = i & 1;

#ifdef HAVE_ASYNC_LDS
    if (i + 1 < nk) stage(cur ^ 1, (i + 1) << 5);   // fire-and-forget to LDS
#else
    if (i + 1 < nk) fetch((i + 1) << 5);            // loads in flight
#endif

    // A fragments for this wave's two 16-row subtiles
    union Frag { v16h v; uint4 q[2]; };
    Frag af[2];
#pragma unroll
    for (int mt = 0; mt < 2; ++mt) {
      const int ar = (wm * 32 + mt * 16 + laneM) * 40 + kh * 8;
      af[mt].q[0] = *(const uint4*)&As[cur][ar];        // K = kh*8 ..
      af[mt].q[1] = *(const uint4*)&As[cur][ar + 16];   // K = 16+kh*8 ..
    }
#pragma unroll
    for (int nt = 0; nt < 4; ++nt) {
      Frag bf;
      const int br = (wn * 64 + nt * 16 + laneM) * 40 + kh * 8;
      bf.q[0] = *(const uint4*)&Bs[cur][br];
      bf.q[1] = *(const uint4*)&Bs[cur][br + 16];
#pragma unroll
      for (int mt = 0; mt < 2; ++mt)
        acc[mt][nt] = __builtin_amdgcn_wmma_f32_16x16x32_f16(
            false, af[mt].v, false, bf.v, (short)0, acc[mt][nt], false, false);
    }

#ifdef HAVE_ASYNC_LDS
    wait_async0();          // next tile fully landed in LDS (after compute)
#else
    if (i + 1 < nk) store(cur ^ 1);   // loadcnt wait happens here, post-WMMA
#endif
    __syncthreads();
  }

  // C/D layout: col = lane&15 (+n tile), rows = base + kh*8 + vgpr index
#pragma unroll
  for (int nt = 0; nt < 4; ++nt) {
    const int col = n0 + wn * 64 + nt * 16 + laneM;
    if (col >= N) continue;
    const float bv = bias ? bias[col] : 0.f;
#pragma unroll
    for (int mt = 0; mt < 2; ++mt) {
      const int row = m0 + wm * 32 + mt * 16 + kh * 8;
#pragma unroll
      for (int v = 0; v < 8; ++v)
        Cout[(size_t)(row + v) * ldc + col] = acc[mt][nt][v] + bv;
    }
  }
}

// ---------------------------------------------------------------------------
// GLU gate + softmax over K=7 conv logits. One thread per (row m, head h).
// ---------------------------------------------------------------------------
__global__ void glu_softmax(const float* __restrict__ y1,
                            const float* __restrict__ glu_b,
                            float* __restrict__ gate,
                            float* __restrict__ wsm) {
  int g = blockIdx.x * blockDim.x + threadIdx.x;
  if (g >= M_DIM * H_DIM) return;
  int m = g >> 4;
  int h = g & 15;
  const float* yr = y1 + (size_t)m * NCAT;

  float a = yr[2 * h]     + glu_b[2 * h];
  float b = yr[2 * h + 1] + glu_b[2 * h + 1];
  gate[g] = a * (1.f / (1.f + __expf(-b)));

  float l[K_CONV];
  float mx = -1e30f;
#pragma unroll
  for (int k = 0; k < K_CONV; ++k) {
    l[k] = yr[32 + h * K_CONV + k];
    mx = fmaxf(mx, l[k]);
  }
  float s = 0.f;
#pragma unroll
  for (int k = 0; k < K_CONV; ++k) { l[k] = __expf(l[k] - mx); s += l[k]; }
  float inv = 1.f / s;
#pragma unroll
  for (int k = 0; k < K_CONV; ++k)
    wsm[(size_t)g * K_CONV + k] = l[k] * inv;
}

// ---------------------------------------------------------------------------
// Depthwise dynamic conv + gate, output f16 for final WMMA GEMM.
// One block per (t,b) row; 4 channels per thread (one head per thread).
// ---------------------------------------------------------------------------
__global__ void __launch_bounds__(256)
dynconv(const float* __restrict__ x,
        const float* __restrict__ gate,
        const float* __restrict__ wsm,
        const int* __restrict__ pad_l_p,
        _Float16* __restrict__ zb) {
  const int m   = blockIdx.x;
  const int tid = threadIdx.x;
  const int t = m >> 3;      // B_DIM == 8
  const int b = m & 7;
  const int pad = *pad_l_p;

  const int c0 = tid * 4;
  const int h  = c0 >> 6;    // 64 channels per head
  const float g = gate[(size_t)m * H_DIM + h];

  float w[K_CONV];
#pragma unroll
  for (int k = 0; k < K_CONV; ++k)
    w[k] = wsm[((size_t)m * H_DIM + h) * K_CONV + k];

  float acc0 = 0.f, acc1 = 0.f, acc2 = 0.f, acc3 = 0.f;
#pragma unroll
  for (int k = 0; k < K_CONV; ++k) {
    const int tt = t + k - pad;
    if (tt < 0 || tt >= T_DIM) continue;
    const float4 xv =
        *(const float4*)&x[((size_t)tt * B_DIM + b) * C_DIM + c0];
    acc0 += w[k] * xv.x; acc1 += w[k] * xv.y;
    acc2 += w[k] * xv.z; acc3 += w[k] * xv.w;
  }
  union { _Float16 h4[4]; uint2 u; } o;
  o.h4[0] = (_Float16)(acc0 * g); o.h4[1] = (_Float16)(acc1 * g);
  o.h4[2] = (_Float16)(acc2 * g); o.h4[3] = (_Float16)(acc3 * g);
  *(uint2*)&zb[(size_t)m * C_DIM + c0] = o.u;
}

// ---------------------------------------------------------------------------
extern "C" void kernel_launch(void* const* d_in, const int* in_sizes, int n_in,
                              void* d_out, int out_size, void* d_ws, size_t ws_size,
                              hipStream_t stream) {
  const float* x        = (const float*)d_in[0];
  const float* w_weight = (const float*)d_in[1];
  const float* glu_w    = (const float*)d_in[2];
  const float* glu_b    = (const float*)d_in[3];
  const float* out_w    = (const float*)d_in[4];
  const float* out_b    = (const float*)d_in[5];
  const int*   pad_l    = (const int*)d_in[6];
  float* out = (float*)d_out;

  char* ws = (char*)d_ws;
  size_t off = 0;
  _Float16* xb   = (_Float16*)(ws + off); off += (size_t)M_DIM * C_DIM * 2;     // 32 MB
  _Float16* zb   = (_Float16*)(ws + off); off += (size_t)M_DIM * C_DIM * 2;     // 32 MB
  _Float16* wcat = (_Float16*)(ws + off); off += (size_t)256 * C_DIM * 2;       // 512 KB (padded)
  _Float16* owb  = (_Float16*)(ws + off); off += (size_t)C_DIM * C_DIM * 2;     // 2 MB
  float*    y1   = (float*)   (ws + off); off += (size_t)M_DIM * NCAT * 4;      // 9 MB
  float*    gate = (float*)   (ws + off); off += (size_t)M_DIM * H_DIM * 4;     // 1 MB
  float*    wsm  = (float*)   (ws + off); off += (size_t)M_DIM * H_DIM * K_CONV * 4; // 7 MB

  // 1) precision conversions / weight packing
  cvt_f32_to_f16<<<(M_DIM * C_DIM / 4 + 255) / 256, 256, 0, stream>>>(x, xb, M_DIM * C_DIM / 4);
  cvt_f32_to_f16<<<(C_DIM * C_DIM / 4 + 255) / 256, 256, 0, stream>>>(out_w, owb, C_DIM * C_DIM / 4);
  prep_wcat<<<(256 * C_DIM / 4 + 255) / 256, 256, 0, stream>>>(glu_w, w_weight, wcat);

  // 2) Y1 = x @ [glu_w ; w_weight]^T   (16384 x 144, K=1024)
  gemm_f16<<<dim3(2, M_DIM / 128), 256, 0, stream>>>(xb, wcat, y1, nullptr, NCAT, C_DIM, NCAT);

  // 3) gate + softmax weights
  glu_softmax<<<(M_DIM * H_DIM + 255) / 256, 256, 0, stream>>>(y1, glu_b, gate, wsm);

  // 4) dynamic depthwise conv + gating -> z (f16)
  dynconv<<<M_DIM, 256, 0, stream>>>(x, gate, wsm, pad_l, zb);

  // 5) out = z @ out_w^T + out_b   (16384 x 1024, K=1024)
  gemm_f16<<<dim3(C_DIM / 128, M_DIM / 128), 256, 0, stream>>>(zb, owb, out, out_b, C_DIM, C_DIM, C_DIM);
}
